// AttentionBlockOld_4020089389408
// MI455X (gfx1250) — matrix operand order repaired and checked by reading, not compiled, not run on hardware
//
#include <hip/hip_runtime.h>
#include <hip/hip_bf16.h>

// ---------------------------------------------------------------------------
// Problem constants (from reference)
// ---------------------------------------------------------------------------
#define B_ 4
#define N_ 2048
#define DIN_ 256
#define H_ 8
#define DHEAD_ 64
#define DQ_ 512            // H*DHEAD
#define DV_ 2048           // H*DIN
#define DQKV_ 3072         // DQ+DQ+DV
#define DEXP_ 1024
#define ROWS_ (B_*N_)      // 8192
#define EPS_ 1e-5f
#define LOG2E_ 1.4426950408889634f

typedef __attribute__((ext_vector_type(16))) _Float16 v16h;
typedef __attribute__((ext_vector_type(8)))  _Float16 v8h;
typedef __attribute__((ext_vector_type(4)))  _Float16 v4h;
typedef __attribute__((ext_vector_type(8)))  float    v8f;
typedef __attribute__((ext_vector_type(4)))  int      v4i;

// ---------------------------------------------------------------------------
// CDNA5 async global->LDS copy (ASYNCcnt path), guarded; batched fallback.
// Builtin signature (from hipcc diagnostic): param0 = v4i in AS1 (__device__),
// param1 = v4i in AS3 (__shared__), then imm offset, imm cpol.
// ---------------------------------------------------------------------------
#if __has_builtin(__builtin_amdgcn_global_load_async_to_lds_b128) && \
    __has_builtin(__builtin_amdgcn_s_wait_asynccnt)
#define HAVE_ASYNC_LDS 1
#else
#define HAVE_ASYNC_LDS 0
#endif

__device__ inline void async_copy_b128(void* lds, const void* g) {
#if HAVE_ASYNC_LDS
  v4i __attribute__((address_space(1)))* gp =
      (v4i __attribute__((address_space(1)))*)(uintptr_t)g;
  v4i __attribute__((address_space(3)))* lp =
      (v4i __attribute__((address_space(3)))*)(uint32_t)(uintptr_t)lds;
  __builtin_amdgcn_global_load_async_to_lds_b128(gp, lp, 0, 0);
#else
  *(v8h*)lds = *(const v8h*)g;
#endif
}
__device__ inline void async_fence() {
#if HAVE_ASYNC_LDS
  __builtin_amdgcn_s_wait_asynccnt(0);
#endif
}

// ---------------------------------------------------------------------------
// WMMA helpers: fragment loads, contiguous k = 16*(lane/16) + element in A and B (wave32)
// ---------------------------------------------------------------------------
__device__ inline v8f wmma_f16(v16h a, v16h b, v8f c) {
  return __builtin_amdgcn_wmma_f32_16x16x32_f16(
      /*neg_a=*/false, a, /*neg_b=*/false, b,
      /*c_mod=*/(short)0, c, /*reuse_a=*/false, /*reuse_b=*/false);
}

// A fragment, 16(M) x 32(K) f16, row-major src (lda halves).
__device__ inline v16h load_frag_a(const _Float16* src, int lda) {
  const int lane = threadIdx.x & 31;
  const int m  = lane & 15;
  const int lh = lane >> 4;
  v16h f;
#pragma unroll
  for (int v = 0; v < 8; ++v) {
    const int k0 = lh * 16 + 2 * v;
    f[2 * v]     = src[m * lda + k0];
    f[2 * v + 1] = src[m * lda + k0 + 1];
  }
  return f;
}

// B fragment (32K x 16N) from a source stored N-major: B(k,n) = src[n*ld + k].
// Per lane this reads 16 contiguous halves -> merges into 2x ds_load_b128.
__device__ inline v16h load_frag_bT(const _Float16* src, int ld) {
  const int lane = threadIdx.x & 31;
  const int n  = lane & 15;
  const int lh = lane >> 4;
  v16h f;
#pragma unroll
  for (int v = 0; v < 8; ++v) {
    const int k0 = lh * 16 + 2 * v;
    f[2 * v]     = src[n * ld + k0];
    f[2 * v + 1] = src[n * ld + k0 + 1];
  }
  return f;
}

// B fragment (32K x 16N) from row-major K-major src (ldb halves/row).
__device__ inline v16h load_frag_b(const _Float16* src, int ldb) {
  const int lane = threadIdx.x & 31;
  const int n  = lane & 15;
  const int lh = lane >> 4;
  v16h f;
#pragma unroll
  for (int v = 0; v < 8; ++v) {
    const int k0 = lh * 16 + 2 * v;
    f[2 * v]     = src[k0 * ldb + n];
    f[2 * v + 1] = src[(k0 + 1) * ldb + n];
  }
  return f;
}

// ---------------------------------------------------------------------------
// 16-lane row reductions via DPP (pure VALU, no LDS): xor masks {1,2,7,15}
// generate the 4-bit group, so the butterfly covers all 16 lanes of a row.
// ---------------------------------------------------------------------------
#if __has_builtin(__builtin_amdgcn_update_dpp)
template <int CTRL>
__device__ inline float dpp_mov(float x) {
  const int xi = __builtin_bit_cast(int, x);
  const int r  = __builtin_amdgcn_update_dpp(xi, xi, CTRL, 0xF, 0xF, true);
  return __builtin_bit_cast(float, r);
}
__device__ inline float rowmax16(float x) {
  x = fmaxf(x, dpp_mov<0xB1>(x));   // quad_perm(1,0,3,2)  : xor 1
  x = fmaxf(x, dpp_mov<0x4E>(x));   // quad_perm(2,3,0,1)  : xor 2
  x = fmaxf(x, dpp_mov<0x141>(x));  // row_half_mirror     : xor 7
  x = fmaxf(x, dpp_mov<0x140>(x));  // row_mirror          : xor 15
  return x;
}
__device__ inline float rowsum16(float x) {
  x += dpp_mov<0xB1>(x);
  x += dpp_mov<0x4E>(x);
  x += dpp_mov<0x141>(x);
  x += dpp_mov<0x140>(x);
  return x;
}
#else
__device__ inline float rowmax16(float x) {
#pragma unroll
  for (int m = 1; m < 16; m <<= 1) x = fmaxf(x, __shfl_xor(x, m, 32));
  return x;
}
__device__ inline float rowsum16(float x) {
#pragma unroll
  for (int m = 1; m < 16; m <<= 1) x += __shfl_xor(x, m, 32);
  return x;
}
#endif

// ---------------------------------------------------------------------------
// fp32 -> f16 weight conversion
// ---------------------------------------------------------------------------
__global__ void cvt_f32_f16(const float* __restrict__ src,
                            _Float16* __restrict__ dst, int n) {
  int i = blockIdx.x * 256 + threadIdx.x;
  if (i < n) dst[i] = (_Float16)src[i];
}

// ---------------------------------------------------------------------------
// FeatureNorm (+optional sinusoidal PE) + Swish, fp32 in -> f16 out.
// ---------------------------------------------------------------------------
template <bool PE>
__global__ void norm_swish(const float* __restrict__ xin,
                           const float* __restrict__ w,
                           const float* __restrict__ bparm,
                           _Float16* __restrict__ outp) {
  const int row = blockIdx.x;        // 0..B*N-1
  const int n   = row % N_;          // token index
  const int d   = threadIdx.x;       // feature index
  float v = xin[(size_t)row * DIN_ + d];
  if (PE) {
    const int j = d >> 1;
    const float div = __expf(-9.210340371976184f * (float)(2 * j) * (1.0f / DIN_));
    const float ang = (float)n * div;
    v += (d & 1) ? cosf(ang) : sinf(ang);
  }
  __shared__ float red[256];
  red[d] = v;
  __syncthreads();
#pragma unroll
  for (int s = 128; s > 0; s >>= 1) {
    if (d < s) red[d] += red[d + s];
    __syncthreads();
  }
  const float mu = red[0] * (1.0f / DIN_);
  __syncthreads();
  const float dv = v - mu;
  red[d] = dv * dv;
  __syncthreads();
#pragma unroll
  for (int s = 128; s > 0; s >>= 1) {
    if (d < s) red[d] += red[d + s];
    __syncthreads();
  }
  const float var = red[0] * (1.0f / DIN_);
  float y = dv * rsqrtf(var + EPS_) * w[d] + bparm[d];
  y = y / (1.0f + __expf(-y));   // swish
  outp[(size_t)row * DIN_ + d] = (_Float16)y;
}

// ---------------------------------------------------------------------------
// Tiled WMMA GEMM: C[M,N] = A[M,K](f16) * B[K,N](f16) + bias, epilogues:
//   EPI==0 : +bias               -> f16 out
//   EPI==1 : swish(+bias)        -> f16 out
//   EPI==2 : +bias +resid (fp32) -> fp32 out
// Block = 256 threads (8 waves), tile 64x64, K-step 32, async LDS staging.
// ---------------------------------------------------------------------------
template <int EPI>
__global__ void __launch_bounds__(256, 2)
gemm16(const _Float16* __restrict__ A, const _Float16* __restrict__ Bw,
       const float* __restrict__ bias, const float* __restrict__ resid,
       _Float16* __restrict__ outH, float* __restrict__ outF,
       int M, int N, int K) {
  __shared__ __align__(16) _Float16 As[64 * 48];  // 64 rows x 32 (pad->48)
  __shared__ __align__(16) _Float16 Bs[32 * 72];  // 32 rows x 64 (pad->72)

  const int tid  = threadIdx.x;
  const int wave = tid >> 5;
  const int lane = tid & 31;
  const int nl   = lane & 15;
  const int lh   = lane >> 4;
  const int wm   = wave & 3;   // 4 M-subtiles of 16
  const int wn   = wave >> 2;  // 2 N-subtiles of 32
  const int am   = blockIdx.y * 64;
  const int bn   = blockIdx.x * 64;

  const int arow = tid >> 2, acc_ = tid & 3;   // A: 1 chunk / thread
  const int brow = tid >> 3, bcc  = tid & 7;   // B: 1 chunk / thread

  v8f acc0 = {0.f, 0.f, 0.f, 0.f, 0.f, 0.f, 0.f, 0.f};
  v8f acc1 = {0.f, 0.f, 0.f, 0.f, 0.f, 0.f, 0.f, 0.f};

  for (int k0 = 0; k0 < K; k0 += 32) {
    __syncthreads();
#if HAVE_ASYNC_LDS
    async_copy_b128(&As[arow * 48 + acc_ * 8],
                    &A[(size_t)(am + arow) * K + k0 + acc_ * 8]);
    async_copy_b128(&Bs[brow * 72 + bcc * 8],
                    &Bw[(size_t)(k0 + brow) * N + bn + bcc * 8]);
    async_fence();
#else
    const v8h ta = *(const v8h*)&A[(size_t)(am + arow) * K + k0 + acc_ * 8];
    const v8h tb = *(const v8h*)&Bw[(size_t)(k0 + brow) * N + bn + bcc * 8];
    *(v8h*)&As[arow * 48 + acc_ * 8] = ta;
    *(v8h*)&Bs[brow * 72 + bcc * 8] = tb;
#endif
    __syncthreads();
    const v16h af = load_frag_a(&As[wm * 16 * 48], 48);
    const v16h b0 = load_frag_b(&Bs[wn * 32], 72);
    const v16h b1 = load_frag_b(&Bs[wn * 32 + 16], 72);
    acc0 = wmma_f16(af, b0, acc0);
    acc1 = wmma_f16(af, b1, acc1);
  }

#pragma unroll
  for (int r = 0; r < 8; ++r) {
    const int row = am + wm * 16 + r + 8 * lh;
    const int c0  = bn + wn * 32 + nl;
    const int c1  = c0 + 16;
    float v0 = acc0[r] + bias[c0];
    float v1 = acc1[r] + bias[c1];
    if (EPI == 1) {
      v0 = v0 / (1.0f + __expf(-v0));
      v1 = v1 / (1.0f + __expf(-v1));
    }
    if (EPI == 2) {
      v0 += resid[(size_t)row * N + c0];
      v1 += resid[(size_t)row * N + c1];
      outF[(size_t)row * N + c0] = v0;
      outF[(size_t)row * N + c1] = v1;
    } else {
      outH[(size_t)row * N + c0] = (_Float16)v0;
      outH[(size_t)row * N + c1] = (_Float16)v1;
    }
  }
}

// ---------------------------------------------------------------------------
// k2[b,h,n] = sum_c k[b,n,h,c]^2  (only k2 needed: q2 cancels in softmax)
// ---------------------------------------------------------------------------
__global__ void k2_kernel(const _Float16* __restrict__ qkv,
                          float* __restrict__ k2) {
  const int idx = blockIdx.x * 256 + threadIdx.x;  // (b*N)*H
  if (idx >= ROWS_ * H_) return;
  const int h = idx & (H_ - 1);
  const int t = idx >> 3;  // b*N + n
  const _Float16* kp = qkv + (size_t)t * DQKV_ + DQ_ + h * DHEAD_;
  float s = 0.f;
#pragma unroll
  for (int u = 0; u < 8; ++u) {
    const v8h kv = *(const v8h*)(kp + u * 8);
#pragma unroll
    for (int e = 0; e < 8; ++e) {
      const float f = (float)kv[e];
      s = fmaf(f, f, s);
    }
  }
  const int b = t / N_;
  const int n = t - b * N_;
  k2[((size_t)b * H_ + h) * N_ + n] = s;
}

// ---------------------------------------------------------------------------
// Flash-style distance-softmax attention.
//   score(n,m) = (2*q.k - k2[m]) / scale^2   (q2 row-constant cancels)
// Grid: (N/128, H, B). Block 256 = 8 waves; wave w owns query tile of 16 rows
// and all 256 V columns of head h (16 f32 WMMA accumulators).
// K tile staged row-major (async); V tile staged TRANSPOSED [col][token] so
// P.V B-fragments read 16 contiguous halves (2x ds_load_b128 each).
// ---------------------------------------------------------------------------
__global__ void __launch_bounds__(256, 1)
attn_kernel(const _Float16* __restrict__ qkv, const float* __restrict__ k2buf,
            _Float16* __restrict__ aout, const float* __restrict__ scale_ptr) {
  __shared__ __align__(16) _Float16 Ks[32 * 64];     // 32 tokens x 64 ch
  __shared__ __align__(16) _Float16 VsT[256 * 32];   // 256 cols x 32 tokens
  __shared__ __align__(16) _Float16 Ps[8 * 16 * 32]; // per-wave P staging

  const int tid  = threadIdx.x;
  const int wave = tid >> 5;
  const int lane = tid & 31;
  const int nl   = lane & 15;
  const int lh   = lane >> 4;
  const int b = blockIdx.z;
  const int h = blockIdx.y;
  const int q0 = (blockIdx.x * 8 + wave) * 16;

  const float sc     = scale_ptr[0];
  const float inv_s2 = 1.0f / (sc * sc);
  const float two_s2 = 2.0f * inv_s2;

  // Q tile 16x64 as two A fragments (K = 0..31, 32..63), loaded once.
  const _Float16* qptr = qkv + (size_t)(b * N_ + q0) * DQKV_ + h * DHEAD_;
  const v16h a0 = load_frag_a(qptr, DQKV_);
  const v16h a1 = load_frag_a(qptr + 32, DQKV_);

  v8f acc[16];
#pragma unroll
  for (int c = 0; c < 16; ++c)
    acc[c] = (v8f){0.f, 0.f, 0.f, 0.f, 0.f, 0.f, 0.f, 0.f};
  float m_run[8], l_run[8];
#pragma unroll
  for (int r = 0; r < 8; ++r) { m_run[r] = -INFINITY; l_run[r] = 0.f; }

  _Float16* pw = &Ps[wave * 16 * 32];
  const float* k2p = k2buf + ((size_t)b * H_ + h) * N_;

  // staging decompositions
  const int krow = tid >> 3, kcc = tid & 7;  // K tile: 1x b128 / thread
  const int vtg  = tid >> 6;                 // V: tokens vtg*8..+8
  const int vcg  = tid & 63;                 // V: cols  vcg*4..+4

  for (int kk = 0; kk < N_; kk += 32) {
    // k2 for this tile: issue early, consumed after the WMMAs.
    const float k2s0 = k2p[kk + nl] * inv_s2;
    const float k2s1 = k2p[kk + 16 + nl] * inv_s2;

    __syncthreads();
    // K tile: 32 tokens x 64 ch (async direct-to-LDS when available)
    async_copy_b128(&Ks[krow * 64 + kcc * 8],
                    &qkv[(size_t)(b * N_ + kk + krow) * DQKV_ + DQ_ +
                         h * DHEAD_ + kcc * 8]);
    // V tile transposed: 8 tokens x 4 cols per thread, register transpose.
    {
      const _Float16* vsrc = &qkv[(size_t)(b * N_ + kk + vtg * 8) * DQKV_ +
                                  2 * DQ_ + h * DIN_ + vcg * 4];
      v4h rr[8];
#pragma unroll
      for (int t = 0; t < 8; ++t)
        rr[t] = *(const v4h*)(vsrc + (size_t)t * DQKV_);
#pragma unroll
      for (int j = 0; j < 4; ++j) {
        v8h col;
#pragma unroll
        for (int t = 0; t < 8; ++t) col[t] = rr[t][j];
        *(v8h*)&VsT[(vcg * 4 + j) * 32 + vtg * 8] = col;
      }
    }
    async_fence();
    __syncthreads();

    // S tiles: 16 q-rows x 2x16 k-tokens, K=64 over head dim (2 WMMA each)
    v8f s0 = (v8f){0.f, 0.f, 0.f, 0.f, 0.f, 0.f, 0.f, 0.f};
    v8f s1 = (v8f){0.f, 0.f, 0.f, 0.f, 0.f, 0.f, 0.f, 0.f};
    s0 = wmma_f16(a0, load_frag_bT(&Ks[0], 64), s0);
    s0 = wmma_f16(a1, load_frag_bT(&Ks[32], 64), s0);
    s1 = wmma_f16(a0, load_frag_bT(&Ks[16 * 64], 64), s1);
    s1 = wmma_f16(a1, load_frag_bT(&Ks[16 * 64 + 32], 64), s1);

    float p0[8], p1[8];
#pragma unroll
    for (int r = 0; r < 8; ++r) {
      const float sc0 = fmaf(s0[r], two_s2, -k2s0);
      const float sc1 = fmaf(s1[r], two_s2, -k2s1);
      const float mt   = fmaxf(rowmax16(sc0), rowmax16(sc1));
      const float mnew = fmaxf(m_run[r], mt);
      const float al   = exp2f((m_run[r] - mnew) * LOG2E_);
      const float e0   = exp2f((sc0 - mnew) * LOG2E_);
      const float e1   = exp2f((sc1 - mnew) * LOG2E_);
      l_run[r] = l_run[r] * al + rowsum16(e0) + rowsum16(e1);
      m_run[r] = mnew;
      p0[r] = e0;
      p1[r] = e1;
#pragma unroll
      for (int c = 0; c < 16; ++c) acc[c][r] *= al;
    }

    // Re-lay P (C layout) into A-fragment layout via per-wave LDS bounce
#pragma unroll
    for (int r = 0; r < 8; ++r) {
      pw[(r + 8 * lh) * 32 + nl]      = (_Float16)p0[r];
      pw[(r + 8 * lh) * 32 + 16 + nl] = (_Float16)p1[r];
    }
    __syncthreads();
    const v16h pa = load_frag_a(pw, 32);
#pragma unroll
    for (int c = 0; c < 16; ++c) {
      const v16h vb = load_frag_bT(&VsT[c * 16 * 32], 32);
      acc[c] = wmma_f16(pa, vb, acc[c]);
    }
  }

  // Normalize and store: out[b, n, h*256 + col] (f16)
#pragma unroll
  for (int r = 0; r < 8; ++r) {
    const float invl = 1.0f / l_run[r];
    const int row = q0 + r + 8 * lh;
    _Float16* op = aout + (size_t)(b * N_ + row) * DV_ + h * DIN_;
#pragma unroll
    for (int c = 0; c < 16; ++c)
      op[c * 16 + nl] = (_Float16)(acc[c][r] * invl);
  }
}

// ---------------------------------------------------------------------------
// Host side
// ---------------------------------------------------------------------------
static inline void* wsoff(void* ws, size_t& off, size_t bytes) {
  void* p = (char*)ws + off;
  off += (bytes + 255) & ~(size_t)255;
  return p;
}

extern "C" void kernel_launch(void* const* d_in, const int* in_sizes, int n_in,
                              void* d_out, int out_size, void* d_ws,
                              size_t ws_size, hipStream_t stream) {
  const float* x       = (const float*)d_in[0];
  const float* scale   = (const float*)d_in[1];
  const float* fn1_w   = (const float*)d_in[2];
  const float* fn1_b   = (const float*)d_in[3];
  const float* qkv_w   = (const float*)d_in[4];
  const float* qkv_b   = (const float*)d_in[5];
  const float* merge_w = (const float*)d_in[6];
  const float* merge_b = (const float*)d_in[7];
  const float* fn2_w   = (const float*)d_in[8];
  const float* fn2_b   = (const float*)d_in[9];
  const float* ff1_w   = (const float*)d_in[10];
  const float* ff1_b   = (const float*)d_in[11];
  const float* ff2_w   = (const float*)d_in[12];
  const float* ff2_b   = (const float*)d_in[13];
  float* out = (float*)d_out;

  size_t off = 0;
  _Float16* qkvW16  = (_Float16*)wsoff(d_ws, off, (size_t)DIN_ * DQKV_ * 2);
  _Float16* mrgW16  = (_Float16*)wsoff(d_ws, off, (size_t)DV_ * DIN_ * 2);
  _Float16* ff1W16  = (_Float16*)wsoff(d_ws, off, (size_t)DIN_ * DEXP_ * 2);
  _Float16* ff2W16  = (_Float16*)wsoff(d_ws, off, (size_t)DEXP_ * DIN_ * 2);
  _Float16* h16     = (_Float16*)wsoff(d_ws, off, (size_t)ROWS_ * DIN_ * 2);
  _Float16* qkv16   = (_Float16*)wsoff(d_ws, off, (size_t)ROWS_ * DQKV_ * 2);
  float*    k2      = (float*)   wsoff(d_ws, off, (size_t)B_ * H_ * N_ * 4);
  _Float16* attn16  = (_Float16*)wsoff(d_ws, off, (size_t)ROWS_ * DV_ * 2);
  float*    x1      = (float*)   wsoff(d_ws, off, (size_t)ROWS_ * DIN_ * 4);
  _Float16* g16     = (_Float16*)wsoff(d_ws, off, (size_t)ROWS_ * DIN_ * 2);
  _Float16* mid16   = (_Float16*)wsoff(d_ws, off, (size_t)ROWS_ * DEXP_ * 2);

  // 1. Convert weights to f16
  cvt_f32_f16<<<(DIN_ * DQKV_ + 255) / 256, 256, 0, stream>>>(qkv_w, qkvW16, DIN_ * DQKV_);
  cvt_f32_f16<<<(DV_ * DIN_ + 255) / 256, 256, 0, stream>>>(merge_w, mrgW16, DV_ * DIN_);
  cvt_f32_f16<<<(DIN_ * DEXP_ + 255) / 256, 256, 0, stream>>>(ff1_w, ff1W16, DIN_ * DEXP_);
  cvt_f32_f16<<<(DEXP_ * DIN_ + 255) / 256, 256, 0, stream>>>(ff2_w, ff2W16, DEXP_ * DIN_);

  // 2. h = swish(featurenorm(x + PE))  (f16)
  norm_swish<true><<<ROWS_, 256, 0, stream>>>(x, fn1_w, fn1_b, h16);

  // 3. qkv = h @ qkv_w + qkv_b  (f16, [8192,3072])
  gemm16<0><<<dim3(DQKV_ / 64, ROWS_ / 64), 256, 0, stream>>>(
      h16, qkvW16, qkv_b, nullptr, qkv16, nullptr, ROWS_, DQKV_, DIN_);

  // 4. k2 row norms
  k2_kernel<<<(ROWS_ * H_ + 255) / 256, 256, 0, stream>>>(qkv16, k2);

  // 5. flash distance-softmax attention -> attn16 [8192, 2048] f16
  attn_kernel<<<dim3(N_ / 128, H_, B_), 256, 0, stream>>>(qkv16, k2, attn16, scale);

  // 6. x1 = x + attn16 @ merge_w + merge_b  (fp32)
  gemm16<2><<<dim3(DIN_ / 64, ROWS_ / 64), 256, 0, stream>>>(
      attn16, mrgW16, merge_b, x, nullptr, x1, ROWS_, DIN_, DV_);

  // 7. g = swish(featurenorm(x1))  (f16)
  norm_swish<false><<<ROWS_, 256, 0, stream>>>(x1, fn2_w, fn2_b, g16);

  // 8. mid = swish(g @ ff1_w + ff1_b)  (f16, [8192,1024])
  gemm16<1><<<dim3(DEXP_ / 64, ROWS_ / 64), 256, 0, stream>>>(
      g16, ff1W16, ff1_b, nullptr, mid16, nullptr, ROWS_, DEXP_, DIN_);

  // 9. out = mid @ ff2_w + ff2_b + x1  (fp32 -> d_out)
  gemm16<2><<<dim3(DIN_ / 64, ROWS_ / 64), 256, 0, stream>>>(
      mid16, ff2W16, ff2_b, x1, nullptr, out, ROWS_, DIN_, DEXP_);

  (void)in_sizes; (void)n_in; (void)out_size; (void)ws_size;
}
